// MultiHeadAttention_64372969833008
// MI455X (gfx1250) — matrix operand order repaired and checked by reading, not compiled, not run on hardware
//
#include <hip/hip_runtime.h>
#include <hip/hip_bf16.h>

typedef __attribute__((ext_vector_type(16))) __bf16 v16bf;
typedef __attribute__((ext_vector_type(8)))  __bf16 v8bf;
typedef __attribute__((ext_vector_type(8)))  float  v8f;

#define NHEADS 16
#define HD     64
#define DMODEL 1024
#define BB     4
#define SS     2048
#define MS     (BB * SS)          // 8192 tokens
#define NEGINF 1e12f

union V16U { v16bf v; v8bf h[2]; };
union V8B  { v8bf  v; __bf16 e[8]; };

// ---------------------------------------------------------------------------
// fp32 -> bf16 elementwise convert (coalesced, 8 elems/thread)
// ---------------------------------------------------------------------------
__global__ __launch_bounds__(256) void cvt_bf16_kernel(const float* __restrict__ X,
                                                       __bf16* __restrict__ Y, int n) {
  int i = (blockIdx.x * 256 + threadIdx.x) * 8;
  if (i >= n) return;
  v8f x = *(const v8f*)(X + i);
  V8B y;
#pragma unroll
  for (int j = 0; j < 8; ++j) y.e[j] = (__bf16)x[j];
  *(v8bf*)(Y + i) = y.v;
}

// ---------------------------------------------------------------------------
// Weight convert + transpose via LDS tile (coalesced both sides):
// Wt[n][k] = (bf16) W[k][n]
// ---------------------------------------------------------------------------
__global__ __launch_bounds__(256) void wt_bf16_kernel(const float* __restrict__ W,
                                                      __bf16* __restrict__ Wt,
                                                      int K, int N) {
  __shared__ float tile[32][33];                    // +1 pad: no bank conflicts
  const int tx = threadIdx.x & 31;
  const int ty = threadIdx.x >> 5;                  // 0..7
  const int nblk = N >> 5;
  const int k0 = (blockIdx.x / nblk) * 32;
  const int n0 = (blockIdx.x % nblk) * 32;
#pragma unroll
  for (int i = 0; i < 32; i += 8)
    tile[ty + i][tx] = W[(size_t)(k0 + ty + i) * N + n0 + tx];
  __syncthreads();
#pragma unroll
  for (int i = 0; i < 32; i += 8)
    Wt[(size_t)(n0 + ty + i) * K + k0 + tx] = (__bf16)tile[tx][ty + i];
}

// ---------------------------------------------------------------------------
// WMMA GEMM: C = (A[M,K] @ Wt^T + bias) * scale   (A bf16, Wt stored [N][K])
// One wave computes a 32x64 tile (2 row-tiles reuse each B fragment).
// MODE 0: f32 row-major; MODE 1: bf16 row-major; MODE 2: bf16 TRANSPOSED
// (C^T[col][row]) stored as packed b128 (8 consecutive rows per fragment).
// ---------------------------------------------------------------------------
template <int MODE>
__global__ __launch_bounds__(256) void gemm_wmma_kernel(const __bf16* __restrict__ Ap,
                                                        const __bf16* __restrict__ Wt,
                                                        const float* __restrict__ bias,
                                                        void* __restrict__ Cp,
                                                        int M, int N, int K, float scale) {
  const int lane = threadIdx.x & 31;
  const int wid  = threadIdx.x >> 5;
  const int gw   = blockIdx.x * 8 + wid;
  const int tn   = N >> 6;
  const int m0   = (gw / tn) * 32;
  const int n0   = (gw % tn) * 64;
  if (m0 >= M) return;
  const int g = lane >> 4;
  const int l = lane & 15;

  v8f acc[2][4] = {};
  const __bf16* a0 = Ap + (size_t)(m0 + l) * K + g * 8;        // row-tile 0
  const __bf16* a1 = a0 + (size_t)16 * K;                      // row-tile 1
  const __bf16* wp = Wt + (size_t)(n0 + l) * K + g * 8;        // B col base

#pragma unroll 2
  for (int k0 = 0; k0 < K; k0 += 32) {
    V16U a[2];
    a[0].h[0] = *(const v8bf*)a0;        a[0].h[1] = *(const v8bf*)(a0 + 16);
    a[1].h[0] = *(const v8bf*)a1;        a[1].h[1] = *(const v8bf*)(a1 + 16);
    a0 += 32; a1 += 32;
#pragma unroll
    for (int nb = 0; nb < 4; ++nb) {
      const __bf16* w = wp + (size_t)(nb * 16) * K;
      V16U b;
      b.h[0] = *(const v8bf*)w;
      b.h[1] = *(const v8bf*)(w + 16);
      acc[0][nb] = __builtin_amdgcn_wmma_f32_16x16x32_bf16(false, a[0].v, false, b.v,
                                                           (short)0, acc[0][nb], false, false);
      acc[1][nb] = __builtin_amdgcn_wmma_f32_16x16x32_bf16(false, a[1].v, false, b.v,
                                                           (short)0, acc[1][nb], false, false);
    }
    wp += 32;
  }

#pragma unroll
  for (int t = 0; t < 2; ++t) {
#pragma unroll
    for (int nb = 0; nb < 4; ++nb) {
      const int col = n0 + nb * 16 + l;
      const float bv = bias[col];
      if (MODE == 2) {
        // C^T[col][row]: rows m0+t*16+8g+r are consecutive -> one packed store
        V8B pk;
#pragma unroll
        for (int r = 0; r < 8; ++r) pk.e[r] = (__bf16)((acc[t][nb][r] + bv) * scale);
        *(v8bf*)((__bf16*)Cp + (size_t)col * M + m0 + t * 16 + 8 * g) = pk.v;
      } else {
#pragma unroll
        for (int r = 0; r < 8; ++r) {
          const int row = m0 + t * 16 + r + 8 * g;
          const float val = (acc[t][nb][r] + bv) * scale;
          if (MODE == 1)
            ((__bf16*)Cp)[(size_t)row * N + col] = (__bf16)val;
          else
            ((float*)Cp)[(size_t)row * N + col] = val;
        }
      }
    }
  }
}

// ---------------------------------------------------------------------------
// Flash attention, transposed-score formulation, no per-iteration barriers.
// Block = one (b, h, 128-row q block); wave = 16 q rows; kv steps of 32.
// S^T = K @ Q^T  -> lane = q row, VGPR = kv  => scalar per-lane softmax state.
// O^T = V^T @ P^T; V^T read directly from the transposed V projection (vwT),
// P^T round-trips per-wave LDS. qw pre-scaled by 1/8; mask = C-initializer.
// ---------------------------------------------------------------------------
__global__ __launch_bounds__(256) void attn_wmma_kernel(const __bf16* __restrict__ qw,
                                                        const __bf16* __restrict__ kw,
                                                        const __bf16* __restrict__ vwT,
                                                        const int* __restrict__ vmask,
                                                        __bf16* __restrict__ ow) {
  __shared__ alignas(32) float  mbias[SS];        // 8 KB: 0 or -1e12 per kv
  __shared__ alignas(32) __bf16 PtS[8][16 * 32];  // 8 KB: P^T per wave [q][kv]

  const int lane = threadIdx.x & 31;
  const int wid  = threadIdx.x >> 5;
  const int QB   = SS / 128;                      // 16 q-blocks per (b,h)
  const int bh   = blockIdx.x / QB;
  const int qblk = blockIdx.x % QB;
  const int bb   = bh / NHEADS;
  const int hh   = bh % NHEADS;
  const int g    = lane >> 4;
  const int l    = lane & 15;
  const int q0   = qblk * 128 + wid * 16;

  // expand mask to additive fp32 bias once per block
  const int* mrow = vmask + (size_t)bb * SS;
  for (int i = threadIdx.x; i < SS; i += 256) mbias[i] = mrow[i] ? 0.0f : -NEGINF;
  __syncthreads();

  // Q^T B-fragments (lane = q col, d contiguous): loaded once
  V16U bq[2];
  {
    const __bf16* qp = qw + (size_t)(bb * SS + q0 + l) * DMODEL + hh * HD + g * 8;
#pragma unroll
    for (int c = 0; c < 2; ++c) {
      bq[c].h[0] = *(const v8bf*)(qp + c * 32);
      bq[c].h[1] = *(const v8bf*)(qp + c * 32 + 16);
    }
  }

  float m = -1e30f, sl = 0.0f;
  v8f o[4] = {};
  __bf16* Pt = PtS[wid];
  const __bf16* kp = kw + (size_t)(bb * SS + l) * DMODEL + hh * HD + g * 8;
  const __bf16* vp = vwT + (size_t)(hh * HD + l) * MS + (size_t)bb * SS + g * 8;
  const float*  mb = mbias + g * 8;

#pragma unroll 1
  for (int t0 = 0; t0 < SS; t0 += 32) {
    // ---- S^T tiles: C initialized with mask bias (kv rows are VGPRs)
    v8f sfr[2];
#pragma unroll
    for (int st = 0; st < 2; ++st) {
      v8f sc = *(const v8f*)(mb + t0 + st * 16);
      const __bf16* kq = kp + (size_t)(st * 16) * DMODEL;
#pragma unroll
      for (int c = 0; c < 2; ++c) {
        V16U ak;
        ak.h[0] = *(const v8bf*)(kq + c * 32);
        ak.h[1] = *(const v8bf*)(kq + c * 32 + 16);
        sc = __builtin_amdgcn_wmma_f32_16x16x32_bf16(false, ak.v, false, bq[c].v,
                                                     (short)0, sc, false, false);
      }
      sfr[st] = sc;
    }
    kp += 32 * DMODEL;

    // ---- online softmax: per-lane scalar state (q row = l)
    float tm = fmaxf(sfr[0][0], sfr[1][0]);
#pragma unroll
    for (int r = 1; r < 8; ++r) tm = fmaxf(tm, fmaxf(sfr[0][r], sfr[1][r]));
    tm = fmaxf(tm, __shfl_xor(tm, 16, 32));
    const float nm    = fmaxf(m, tm);
    const float alpha = __expf(m - nm);
    m = nm;
#pragma unroll
    for (int nb = 0; nb < 4; ++nb) o[nb] *= alpha;

    float rs = 0.0f;
#pragma unroll
    for (int st = 0; st < 2; ++st) {
      V8B pk;
#pragma unroll
      for (int r = 0; r < 8; ++r) {
        const float p = __expf(sfr[st][r] - nm);
        rs += p;
        pk.e[r] = (__bf16)p;
      }
      *(v8bf*)(Pt + l * 32 + st * 16 + 8 * g) = pk.v;   // 8 contiguous kv
    }
    rs += __shfl_xor(rs, 16, 32);
    sl = sl * alpha + rs;

    // ---- O^T += V^T @ P^T (V^T straight from global; DS in-order per wave)
    V16U bp;
    {
      const __bf16* p = Pt + l * 32 + g * 8;
      bp.h[0] = *(const v8bf*)p;
      bp.h[1] = *(const v8bf*)(p + 16);
    }
#pragma unroll
    for (int nb = 0; nb < 4; ++nb) {
      const __bf16* p = vp + (size_t)(nb * 16) * MS + t0;
      V16U av;
      av.h[0] = *(const v8bf*)p;
      av.h[1] = *(const v8bf*)(p + 16);
      o[nb] = __builtin_amdgcn_wmma_f32_16x16x32_bf16(false, av.v, false, bp.v,
                                                      (short)0, o[nb], false, false);
    }
  }

  // ---- normalize and store O^T: element (r,lane) -> (d = nb*16+8g+r, q = q0+l)
  const float inv = 1.0f / sl;
  __bf16* op = ow + (size_t)(bb * SS + q0 + l) * DMODEL + hh * HD;
#pragma unroll
  for (int nb = 0; nb < 4; ++nb) {
    V8B pk;
#pragma unroll
    for (int r = 0; r < 8; ++r) pk.e[r] = (__bf16)(o[nb][r] * inv);
    *(v8bf*)(op + nb * 16 + 8 * g) = pk.v;
  }
}

// ---------------------------------------------------------------------------
// Launch
// ---------------------------------------------------------------------------
extern "C" void kernel_launch(void* const* d_in, const int* in_sizes, int n_in,
                              void* d_out, int out_size, void* d_ws, size_t ws_size,
                              hipStream_t stream) {
  (void)in_sizes; (void)n_in; (void)out_size; (void)ws_size;
  const float* q  = (const float*)d_in[0];
  const float* k  = (const float*)d_in[1];
  const float* v  = (const float*)d_in[2];
  const int*  msk = (const int*) d_in[3];
  const float* Wq = (const float*)d_in[4];
  const float* bq = (const float*)d_in[5];
  const float* Wk = (const float*)d_in[6];
  const float* bk = (const float*)d_in[7];
  const float* Wv = (const float*)d_in[8];
  const float* bv = (const float*)d_in[9];
  const float* Wo = (const float*)d_in[10];
  const float* bo = (const float*)d_in[11];
  float* out = (float*)d_out;

  const size_t NTOK = MS;                    // 8192
  const size_t NACT = NTOK * DMODEL;         // 8M elements
  __bf16* WqT = (__bf16*)d_ws;
  __bf16* WkT = WqT + (size_t)DMODEL * DMODEL;
  __bf16* WvT = WkT + (size_t)DMODEL * DMODEL;
  __bf16* WoT = WvT + (size_t)DMODEL * DMODEL;
  __bf16* qb  = WoT + (size_t)DMODEL * DMODEL;
  __bf16* kb  = qb + NACT;
  __bf16* vb  = kb + NACT;
  __bf16* qw  = vb + NACT;
  __bf16* kw  = qw + NACT;
  __bf16* vwT = kw + NACT;                   // transposed: [H*64][B*S]
  __bf16* ow  = vwT + NACT;

  const dim3 blk(256);

  // activation converts: 8M / (256*8) = 4096 blocks
  cvt_bf16_kernel<<<4096, blk, 0, stream>>>(q, qb, (int)NACT);
  cvt_bf16_kernel<<<4096, blk, 0, stream>>>(k, kb, (int)NACT);
  cvt_bf16_kernel<<<4096, blk, 0, stream>>>(v, vb, (int)NACT);

  // tiled weight transposes: (1024/32)^2 = 1024 blocks
  wt_bf16_kernel<<<1024, blk, 0, stream>>>(Wq, WqT, DMODEL, DMODEL);
  wt_bf16_kernel<<<1024, blk, 0, stream>>>(Wk, WkT, DMODEL, DMODEL);
  wt_bf16_kernel<<<1024, blk, 0, stream>>>(Wv, WvT, DMODEL, DMODEL);
  wt_bf16_kernel<<<1024, blk, 0, stream>>>(Wo, WoT, DMODEL, DMODEL);

  // GEMM waves = (8192/32)*(1024/64) = 4096 -> 512 blocks of 8 waves
  gemm_wmma_kernel<1><<<512, blk, 0, stream>>>(qb, WqT, bq, qw,  MS, DMODEL, DMODEL, 0.125f);
  gemm_wmma_kernel<1><<<512, blk, 0, stream>>>(kb, WkT, bk, kw,  MS, DMODEL, DMODEL, 1.0f);
  gemm_wmma_kernel<2><<<512, blk, 0, stream>>>(vb, WvT, bv, vwT, MS, DMODEL, DMODEL, 1.0f);

  // attention: blocks = B*H*(S/128) = 1024
  attn_wmma_kernel<<<1024, blk, 0, stream>>>(qw, kw, vwT, msk, ow);

  gemm_wmma_kernel<0><<<512, blk, 0, stream>>>(ow, WoT, bo, out, MS, DMODEL, DMODEL, 1.0f);
}
